// EncoderLayer_51110110823018
// MI455X (gfx1250) — matrix operand order repaired and checked by reading, not compiled, not run on hardware
//
#include <hip/hip_runtime.h>
#include <hip/hip_bf16.h>

typedef __attribute__((ext_vector_type(16))) _Float16 v16h;
typedef __attribute__((ext_vector_type(8)))  _Float16 v8h;
typedef __attribute__((ext_vector_type(4)))  _Float16 v4h;
typedef __attribute__((ext_vector_type(8)))  float    v8f;

#define D_MODEL 1024
#define HEADS   16
#define D_FF    4096
#define SEQ     2048

__device__ __forceinline__ v16h join16(v8h lo, v8h hi) {
    return __builtin_shufflevector(lo, hi, 0,1,2,3,4,5,6,7,8,9,10,11,12,13,14,15);
}

__device__ __forceinline__ v8f wmma_f16(v16h a, v16h b, v8f c) {
    // D = A(16x32 f16) * B(32x16 f16) + C(16x16 f32)
    return __builtin_amdgcn_wmma_f32_16x16x32_f16(
        /*neg_a=*/false, a, /*neg_b=*/false, b,
        /*c_mod=*/(short)0, c, /*reuse_a=*/false, /*reuse_b=*/false);
}

// ---------------------------------------------------------------- convert
__global__ __launch_bounds__(256)
void cvt_f32_f16_kernel(const float* __restrict__ src, _Float16* __restrict__ dst, size_t n) {
    size_t i = ((size_t)blockIdx.x * blockDim.x + threadIdx.x) * 4;
    size_t stride = (size_t)gridDim.x * blockDim.x * 4;
    for (; i < n; i += stride) {
        float4 v = *(const float4*)(src + i);
        v4h o;
        o[0] = (_Float16)v.x; o[1] = (_Float16)v.y;
        o[2] = (_Float16)v.z; o[3] = (_Float16)v.w;
        *(v4h*)(dst + i) = o;
    }
}

// ---------------------------------------------------------------- GEMM
// C[M,N] = A[M,K](f16) @ B[K,N](f16) + bias, optional ReLU.
// Block tile 256x64, K-step 32, double-buffered LDS.
// 8 waves; wave w computes rows [w*32, w*32+32) as 2 row-groups x 4 col-tiles.
#define GEMM_BM 256
#define GEMM_BN 64
#define GEMM_BK 32

template <bool WF, bool WH, bool RELU>
__global__ __launch_bounds__(256)
void gemm_wmma_kernel(const _Float16* __restrict__ A, const _Float16* __restrict__ B,
                      const float* __restrict__ bias,
                      float* __restrict__ outF, _Float16* __restrict__ outH,
                      int M, int N, int K)
{
    __shared__ __align__(16) _Float16 As[2][GEMM_BM * GEMM_BK];  // row-major [256][32]
    __shared__ __align__(16) _Float16 Bt[2][GEMM_BN * GEMM_BK];  // transposed [n][k] = [64][32]

    const int tid  = threadIdx.x;
    const int lane = tid & 31;
    const int wave = tid >> 5;
    const int m0   = blockIdx.x * GEMM_BM;
    const int n0   = blockIdx.y * GEMM_BN;

    const int arow = lane & 15;
    const int asel = (lane >> 4) & 1;          // A frag k-half (contiguous k = 16h + s)
    const int kb   = (lane & 16) ? 16 : 0;     // B frag k-base (contiguous k = 16h + s)
    const int bidx = tid * 8;
    const int br   = bidx >> 6;                // B tile row (k)
    const int bc   = bidx & 63;                // B tile col (n)

    v8h areg[4], breg;                         // global prefetch registers
    auto load_tiles = [&](int k0) {
        const _Float16* asrc = A + (size_t)(m0 + tid) * K + k0;   // 1 full A row / thread
        areg[0] = *(const v8h*)(asrc);
        areg[1] = *(const v8h*)(asrc + 8);
        areg[2] = *(const v8h*)(asrc + 16);
        areg[3] = *(const v8h*)(asrc + 24);
        breg    = *(const v8h*)(B + (size_t)(k0 + br) * N + n0 + bc);
    };
    auto store_tiles = [&](int buf) {
        *(v8h*)(&As[buf][tid * 32 + 0])  = areg[0];
        *(v8h*)(&As[buf][tid * 32 + 8])  = areg[1];
        *(v8h*)(&As[buf][tid * 32 + 16]) = areg[2];
        *(v8h*)(&As[buf][tid * 32 + 24]) = areg[3];
#pragma unroll
        for (int i = 0; i < 8; ++i) Bt[buf][(bc + i) * GEMM_BK + br] = breg[i];
    };

    v8f acc[2][4];
#pragma unroll
    for (int r = 0; r < 2; ++r)
#pragma unroll
        for (int j = 0; j < 4; ++j)
#pragma unroll
            for (int i = 0; i < 8; ++i) acc[r][j][i] = 0.0f;

    load_tiles(0);
    store_tiles(0);
    __syncthreads();

    int cur = 0;
    for (int k0 = 0; k0 < K; k0 += GEMM_BK) {
        const bool has_next = (k0 + GEMM_BK) < K;
        if (has_next) load_tiles(k0 + GEMM_BK);   // overlap with compute below

        v16h af[2];
#pragma unroll
        for (int r = 0; r < 2; ++r) {
            const _Float16* ap = &As[cur][(wave * 32 + r * 16 + arow) * GEMM_BK + asel * 16];
            af[r] = join16(*(const v8h*)ap, *(const v8h*)(ap + 8));
        }
#pragma unroll
        for (int j = 0; j < 4; ++j) {
            const _Float16* bp = &Bt[cur][(j * 16 + (lane & 15)) * GEMM_BK + kb];
            v16h bf = join16(*(const v8h*)bp, *(const v8h*)(bp + 8));
            acc[0][j] = wmma_f16(af[0], bf, acc[0][j]);
            acc[1][j] = wmma_f16(af[1], bf, acc[1][j]);
        }

        if (has_next) {
            store_tiles(cur ^ 1);   // other buffer: no barrier needed before stores
            __syncthreads();        // make stores visible for next iteration
            cur ^= 1;
        }
    }

#pragma unroll
    for (int r = 0; r < 2; ++r) {
        const int rbase = m0 + wave * 32 + r * 16 + ((lane & 16) ? 8 : 0);
#pragma unroll
        for (int j = 0; j < 4; ++j) {
            const int col = n0 + j * 16 + (lane & 15);
            const float bi = bias[col];
#pragma unroll
            for (int i = 0; i < 8; ++i) {
                float v = acc[r][j][i] + bi;
                if (RELU) v = fmaxf(v, 0.0f);
                size_t off = (size_t)(rbase + i) * N + col;
                if (WF) outF[off] = v;
                if (WH) outH[off] = (_Float16)v;
            }
        }
    }
}

// ---------------------------------------------------------------- attention
// Block: one (b,h), 128 query rows (8 waves x 16). Flash-style online softmax.
// K/V tiles double-buffered in LDS; P re-layout through per-wave LDS.
// Row-sum of P computed by an extra WMMA against an all-ones B fragment.
__global__ __launch_bounds__(256)
void attn_wmma_kernel(const _Float16* __restrict__ Qh, const _Float16* __restrict__ Kh,
                      const _Float16* __restrict__ Vh, const int* __restrict__ mask,
                      _Float16* __restrict__ ctx, int S, int D, int H)
{
    __shared__ __align__(16) _Float16 Kt[2][32 * 64];    // [key][d]
    __shared__ __align__(16) _Float16 Vt[2][64 * 32];    // [d][key] (transposed)
    __shared__ __align__(16) _Float16 Pw[8][16 * 32];    // per-wave P tile

    const int tid  = threadIdx.x;
    const int lane = tid & 31;
    const int wave = tid >> 5;
    const int bh = blockIdx.y;
    const int b = bh / H, h = bh % H;
    const size_t base = (size_t)b * S * D + (size_t)h * 64;
    const int qrow = blockIdx.x * 128 + wave * 16;

    const float NEG = -__builtin_inff();
    const int arow = lane & 15;
    const int asel = (lane >> 4) & 1;
    const int keyl = lane & 15;
    const int kb   = (lane & 16) ? 16 : 0;
    const int prow = (lane & 16) ? 8 : 0;
    const int lidx = tid * 8;
    const int lr   = lidx >> 6;                 // tile row (key)
    const int lc   = lidx & 63;                 // tile col (d)

    // Q fragments for d-chunks 0 and 1; fold 1/sqrt(dk)=0.125 into Q (exact pow2 in f16)
    v16h qf[2];
#pragma unroll
    for (int c = 0; c < 2; ++c) {
        const _Float16* qp = Qh + base + (size_t)(qrow + arow) * D + c * 32 + asel * 16;
        qf[c] = join16(*(const v8h*)qp, *(const v8h*)(qp + 8));
#pragma unroll
        for (int i = 0; i < 16; ++i) qf[c][i] = qf[c][i] * (_Float16)0.125f;
    }

    v16h ones;
#pragma unroll
    for (int i = 0; i < 16; ++i) ones[i] = (_Float16)1.0f;

    v8h kreg, vreg;
    auto load_kv = [&](int kt) {
        kreg = *(const v8h*)(Kh + base + (size_t)(kt + lr) * D + lc);
        vreg = *(const v8h*)(Vh + base + (size_t)(kt + lr) * D + lc);
    };
    auto store_kv = [&](int buf) {
        *(v8h*)(&Kt[buf][lidx]) = kreg;
#pragma unroll
        for (int i = 0; i < 8; ++i) Vt[buf][(lc + i) * 32 + lr] = vreg[i];
    };

    v8f acc[4], accl;                           // context and row-sum accumulators
    float mrow[8];
#pragma unroll
    for (int i = 0; i < 8; ++i) { mrow[i] = NEG; accl[i] = 0.0f; }
#pragma unroll
    for (int j = 0; j < 4; ++j)
#pragma unroll
        for (int i = 0; i < 8; ++i) acc[j][i] = 0.0f;

    const int* mrowp = mask + (size_t)b * S;

    load_kv(0);
    store_kv(0);
    __syncthreads();

    int cur = 0;
    for (int kt = 0; kt < S; kt += 32) {
        const bool has_next = (kt + 32) < S;
        if (has_next) load_kv(kt + 32);         // overlap with compute below

        // scores: S0 = keys kt..kt+15, S1 = keys kt+16..kt+31 (scale pre-folded)
        v8f s0, s1;
#pragma unroll
        for (int i = 0; i < 8; ++i) { s0[i] = 0.0f; s1[i] = 0.0f; }
#pragma unroll
        for (int c = 0; c < 2; ++c) {
            const _Float16* k0p = &Kt[cur][keyl * 64 + c * 32 + kb];
            const _Float16* k1p = &Kt[cur][(16 + keyl) * 64 + c * 32 + kb];
            v16h bk0 = join16(*(const v8h*)k0p, *(const v8h*)(k0p + 8));
            v16h bk1 = join16(*(const v8h*)k1p, *(const v8h*)(k1p + 8));
            s0 = wmma_f16(qf[c], bk0, s0);
            s1 = wmma_f16(qf[c], bk1, s1);
        }

        // additive mask bias: 0 (keep) or -inf (drop); exp(-inf - m) == 0 naturally
        const float mb0 = (mrowp[kt + keyl]      != 0) ? 0.0f : NEG;
        const float mb1 = (mrowp[kt + 16 + keyl] != 0) ? 0.0f : NEG;

        float pr0[8], pr1[8];
#pragma unroll
        for (int i = 0; i < 8; ++i) {
            float a  = s0[i] + mb0;
            float bb = s1[i] + mb1;
            float tm = fmaxf(a, bb);
#pragma unroll
            for (int off = 1; off < 16; off <<= 1) tm = fmaxf(tm, __shfl_xor(tm, off, 32));
            float mn = fmaxf(mrow[i], tm);
            float mc = fmaxf(mn, -3.0e38f);     // finite exponent ref, NaN-free if all masked
            float resc = __expf(mrow[i] - mc);
            pr0[i] = __expf(a - mc);
            pr1[i] = __expf(bb - mc);
            mrow[i] = mn;
            accl[i] *= resc;
#pragma unroll
            for (int j = 0; j < 4; ++j) acc[j][i] *= resc;
        }

        // Re-layout P (C-frag) -> A-frag through per-wave LDS
        _Float16* pw = &Pw[wave][0];
#pragma unroll
        for (int i = 0; i < 8; ++i) {
            pw[(i + prow) * 32 + keyl]      = (_Float16)pr0[i];
            pw[(i + prow) * 32 + 16 + keyl] = (_Float16)pr1[i];
        }
        asm volatile("s_wait_dscnt 0" ::: "memory");
        const _Float16* pp = &pw[arow * 32 + asel * 16];
        v16h pf = join16(*(const v8h*)pp, *(const v8h*)(pp + 8));

        // row-sum(P) via WMMA against all-ones B: every lane gets l in all 8 slots
        accl = wmma_f16(pf, ones, accl);
#pragma unroll
        for (int j = 0; j < 4; ++j) {
            const _Float16* vp = &Vt[cur][(j * 16 + keyl) * 32 + kb];
            v16h bv = join16(*(const v8h*)vp, *(const v8h*)(vp + 8));
            acc[j] = wmma_f16(pf, bv, acc[j]);
        }

        if (has_next) {
            store_kv(cur ^ 1);      // other buffer: safe without extra barrier
            __syncthreads();
            cur ^= 1;
        }
    }

    // normalize and write ctx (f16)
#pragma unroll
    for (int i = 0; i < 8; ++i) {
        const float invl = accl[i] > 0.0f ? 1.0f / accl[i] : 0.0f;
        const int row = qrow + i + prow;
#pragma unroll
        for (int j = 0; j < 4; ++j) {
            const int col = j * 16 + keyl;
            ctx[base + (size_t)row * D + col] = (_Float16)(acc[j][i] * invl);
        }
    }
}

// ---------------------------------------------------------------- add + layernorm
template <bool WH>
__global__ __launch_bounds__(256)
void add_ln_kernel(const float* __restrict__ A, const float* __restrict__ Bv,
                   const float* __restrict__ g, const float* __restrict__ be,
                   float* __restrict__ outF, _Float16* __restrict__ outH, int D)
{
    const int row = blockIdx.x;
    const float* a = A + (size_t)row * D;
    const float* b = Bv + (size_t)row * D;
    __shared__ float rsum[8], rsq[8];

    float vals[4];
    float sum = 0.0f, sq = 0.0f;
#pragma unroll
    for (int i = 0; i < 4; ++i) {
        int c = threadIdx.x + i * 256;
        float v = a[c] + b[c];
        vals[i] = v; sum += v; sq += v * v;
    }
#pragma unroll
    for (int off = 16; off >= 1; off >>= 1) {
        sum += __shfl_xor(sum, off, 32);
        sq  += __shfl_xor(sq,  off, 32);
    }
    const int wave = threadIdx.x >> 5, lane = threadIdx.x & 31;
    if (lane == 0) { rsum[wave] = sum; rsq[wave] = sq; }
    __syncthreads();
    float ts = 0.0f, tq = 0.0f;
#pragma unroll
    for (int i = 0; i < 8; ++i) { ts += rsum[i]; tq += rsq[i]; }
    const float mu  = ts / (float)D;
    const float var = tq / (float)D - mu * mu;
    const float inv = rsqrtf(var + 1e-5f);
#pragma unroll
    for (int i = 0; i < 4; ++i) {
        int c = threadIdx.x + i * 256;
        float y = (vals[i] - mu) * inv * g[c] + be[c];
        size_t off = (size_t)row * D + c;
        outF[off] = y;
        if (WH) outH[off] = (_Float16)y;
    }
}

// ---------------------------------------------------------------- launch
extern "C" void kernel_launch(void* const* d_in, const int* in_sizes, int n_in,
                              void* d_out, int out_size, void* d_ws, size_t ws_size,
                              hipStream_t stream)
{
    const float* x   = (const float*)d_in[0];
    const int*  mask = (const int*)d_in[1];
    const float* Wq = (const float*)d_in[2];  const float* bq = (const float*)d_in[3];
    const float* Wk = (const float*)d_in[4];  const float* bk = (const float*)d_in[5];
    const float* Wv = (const float*)d_in[6];  const float* bv = (const float*)d_in[7];
    const float* Wo = (const float*)d_in[8];  const float* bo = (const float*)d_in[9];
    const float* W1 = (const float*)d_in[10]; const float* b1 = (const float*)d_in[11];
    const float* W2 = (const float*)d_in[12]; const float* b2 = (const float*)d_in[13];
    const float* g1 = (const float*)d_in[14]; const float* be1 = (const float*)d_in[15];
    const float* g2 = (const float*)d_in[16]; const float* be2 = (const float*)d_in[17];

    const int D = D_MODEL, F = D_FF, H = HEADS, S = SEQ;
    const int M = in_sizes[0] / D;   // B*S = 8192
    const int B = M / S;

    // carve workspace (all offsets 256B-aligned)
    char* p = (char*)d_ws;
    auto carve = [&](size_t bytes) -> char* {
        char* r = p; p += (bytes + 255) & ~(size_t)255; return r;
    };
    _Float16* xh  = (_Float16*)carve((size_t)M * D * 2);
    _Float16* Wqh = (_Float16*)carve((size_t)D * D * 2);
    _Float16* Wkh = (_Float16*)carve((size_t)D * D * 2);
    _Float16* Wvh = (_Float16*)carve((size_t)D * D * 2);
    _Float16* Woh = (_Float16*)carve((size_t)D * D * 2);
    _Float16* W1h = (_Float16*)carve((size_t)D * F * 2);
    _Float16* W2h = (_Float16*)carve((size_t)F * D * 2);
    _Float16* Qh  = (_Float16*)carve((size_t)M * D * 2);
    _Float16* Kh  = (_Float16*)carve((size_t)M * D * 2);
    _Float16* Vh  = (_Float16*)carve((size_t)M * D * 2);
    _Float16* Ch  = (_Float16*)carve((size_t)M * D * 2);
    float*    att = (float*)   carve((size_t)M * D * 4);
    float*    hb  = (float*)   carve((size_t)M * D * 4);
    _Float16* hh  = (_Float16*)carve((size_t)M * D * 2);
    _Float16* ff1 = (_Float16*)carve((size_t)M * F * 2);
    float*    ff2 = (float*)   carve((size_t)M * D * 4);
    (void)ws_size; (void)n_in; (void)out_size;

    // f32 -> f16 conversions
    cvt_f32_f16_kernel<<<1024, 256, 0, stream>>>(x,  xh,  (size_t)M * D);
    cvt_f32_f16_kernel<<<256,  256, 0, stream>>>(Wq, Wqh, (size_t)D * D);
    cvt_f32_f16_kernel<<<256,  256, 0, stream>>>(Wk, Wkh, (size_t)D * D);
    cvt_f32_f16_kernel<<<256,  256, 0, stream>>>(Wv, Wvh, (size_t)D * D);
    cvt_f32_f16_kernel<<<256,  256, 0, stream>>>(Wo, Woh, (size_t)D * D);
    cvt_f32_f16_kernel<<<1024, 256, 0, stream>>>(W1, W1h, (size_t)D * F);
    cvt_f32_f16_kernel<<<1024, 256, 0, stream>>>(W2, W2h, (size_t)F * D);

    // QKV projections (f16 out only)
    dim3 gDD(M / GEMM_BM, D / GEMM_BN);
    gemm_wmma_kernel<false, true, false><<<gDD, 256, 0, stream>>>(xh, Wqh, bq, nullptr, Qh, M, D, D);
    gemm_wmma_kernel<false, true, false><<<gDD, 256, 0, stream>>>(xh, Wkh, bk, nullptr, Kh, M, D, D);
    gemm_wmma_kernel<false, true, false><<<gDD, 256, 0, stream>>>(xh, Wvh, bv, nullptr, Vh, M, D, D);

    // attention
    attn_wmma_kernel<<<dim3(S / 128, B * H), 256, 0, stream>>>(Qh, Kh, Vh, mask, Ch, S, D, H);

    // output projection (f32 out) + residual LN (f32 + f16 out)
    gemm_wmma_kernel<true, false, false><<<gDD, 256, 0, stream>>>(Ch, Woh, bo, att, nullptr, M, D, D);
    add_ln_kernel<true><<<M, 256, 0, stream>>>(x, att, g1, be1, hb, hh, D);

    // FFN
    dim3 gDF(M / GEMM_BM, F / GEMM_BN);
    gemm_wmma_kernel<false, true, true ><<<gDF, 256, 0, stream>>>(hh, W1h, b1, nullptr, ff1, M, F, D);
    gemm_wmma_kernel<true, false, false><<<gDD, 256, 0, stream>>>(ff1, W2h, b2, ff2, nullptr, M, D, F);
    add_ln_kernel<false><<<M, 256, 0, stream>>>(hb, ff2, g2, be2, (float*)d_out, nullptr, D);
}
